// Transformer_topk_36558761624168
// MI455X (gfx1250) — matrix operand order repaired and checked by reading, not compiled, not run on hardware
//
#include <hip/hip_runtime.h>
#include <hip/hip_bf16.h>

// CDNA5 / gfx1250: wave32, WMMA 16x16x32 bf16 -> f32 accumulate.
typedef __bf16 bf16_t;
typedef __attribute__((ext_vector_type(16))) __bf16 v16bf;
typedef __attribute__((ext_vector_type(8)))  float  v8f;
typedef __attribute__((ext_vector_type(4)))  float  v4f;
typedef __attribute__((ext_vector_type(4)))  unsigned int v4u;

#define DIMM   768
#define HEADS  12
#define DHEAD  64
#define MLPD   3072
#define DEPTHL 4
#define BATCH  64
#define NTOK   197
#define LN_EPS 1e-5f

// ---------------------------------------------------------------- utilities
// n must be a multiple of 4 (all tensors here are).
__global__ void k_cvt_bf16(const float* __restrict__ s, bf16_t* __restrict__ d,
                           long long n4) {
  long long i = (long long)blockIdx.x * blockDim.x + threadIdx.x;
  const long long stride = (long long)gridDim.x * blockDim.x;
  for (; i < n4; i += stride) {
    const v4f v = ((const v4f*)s)[i];
    bf16_t* o = d + 4 * i;
    o[0] = (bf16_t)v.x; o[1] = (bf16_t)v.y; o[2] = (bf16_t)v.z; o[3] = (bf16_t)v.w;
  }
}

// LayerNorm over dim=768; one 256-thread block per row; bf16 and/or f32 out.
__global__ void k_layernorm(const float* __restrict__ x, const float* __restrict__ g,
                            const float* __restrict__ b, bf16_t* __restrict__ ob,
                            float* __restrict__ of) {
  const long long row = blockIdx.x;
  const float* xr = x + row * DIMM;
  const int t = threadIdx.x;  // 256
  float v0 = xr[t], v1 = xr[t + 256], v2 = xr[t + 512];
  __shared__ float rs[256];
  rs[t] = v0 + v1 + v2;
  __syncthreads();
  for (int o = 128; o; o >>= 1) { if (t < o) rs[t] += rs[t + o]; __syncthreads(); }
  const float mean = rs[0] * (1.0f / 768.0f);
  __syncthreads();
  const float d0 = v0 - mean, d1 = v1 - mean, d2 = v2 - mean;
  rs[t] = d0 * d0 + d1 * d1 + d2 * d2;
  __syncthreads();
  for (int o = 128; o; o >>= 1) { if (t < o) rs[t] += rs[t + o]; __syncthreads(); }
  const float rstd = rsqrtf(rs[0] * (1.0f / 768.0f) + LN_EPS);
  const float y0 = d0 * rstd * g[t]       + b[t];
  const float y1 = d1 * rstd * g[t + 256] + b[t + 256];
  const float y2 = d2 * rstd * g[t + 512] + b[t + 512];
  if (ob) { bf16_t* o = ob + row * DIMM; o[t] = (bf16_t)y0; o[t+256] = (bf16_t)y1; o[t+512] = (bf16_t)y2; }
  if (of) { float*  o = of + row * DIMM; o[t] = y0;         o[t+256] = y1;         o[t+512] = y2; }
}

// ---------------------------------------------------------------- WMMA GEMM
// C[M,N] = act(A[M,K] @ W[K,N] + bias) (+resid). A,W bf16 row-major.
// Block tile 64x128, K-step 32, double-buffered LDS. 8 waves: each owns a
// 32x32 output tile = 2x2 WMMA accumulators.
// LDS layouts (padded rows = 40 bf16 = 80B, conflict-free, 16B-aligned):
//   sA[buf][m][k]            (row-major, lane fragment = 2x contiguous 16B)
//   sB[buf][n][k]            (transposed + K-pair packed: dword kk holds
//                             (W[2kk][n], W[2kk+1][n]) -> lane fragment is
//                             one contiguous 32B run -> 2x ds_load_b128)
#define BM 64
#define BN 128
#define BKK 32
#define LPAD 8  // row = 40 bf16

__global__ __launch_bounds__(256) void k_gemm_bf16(
    const bf16_t* __restrict__ A, const bf16_t* __restrict__ W,
    const float* __restrict__ bias, const float* __restrict__ resid,
    float* __restrict__ Cf, bf16_t* __restrict__ Cb,
    int M, int N, int K, int gelu) {
  __shared__ __align__(16) bf16_t sA[2][BM][BKK + LPAD];
  __shared__ __align__(16) bf16_t sB[2][BN][BKK + LPAD];
  const int bm = blockIdx.y * BM;
  const int bn = blockIdx.x * BN;
  const int tid  = threadIdx.x;
  const int lane = tid & 31;
  const int wave = tid >> 5;
  const int wy = wave >> 2;   // 0..1
  const int wx = wave & 3;    // 0..3
  const int half = lane >> 4; // K-half select: k = 16*half + s in A and B fragments
  const int l16  = lane & 15;

  const v8f vzero = {0.f, 0.f, 0.f, 0.f, 0.f, 0.f, 0.f, 0.f};
  v8f acc[2][2];
  acc[0][0] = vzero; acc[0][1] = vzero; acc[1][0] = vzero; acc[1][1] = vzero;

  // --- staging registers (whole next tile held in VGPRs) ---
  const int arow_ld = tid >> 2;           // 0..63
  const int acol_ld = (tid & 3) * 8;      // bf16 column, 16B granules
  v4u areg;
  unsigned int b0reg[4], b1reg[4];

  auto ldtile = [&](int kk0) {
    areg = *(const v4u*)(A + (long long)(bm + arow_ld) * K + kk0 + acol_ld);
#pragma unroll
    for (int e = 0; e < 4; e++) {
      const int u = tid + 256 * e;
      const int npair = u & 63;           // n = 2*npair
      const int kk = u >> 6;              // 0..15 (K dword pair)
      const bf16_t* wp = W + (long long)(kk0 + 2 * kk) * N + bn + 2 * npair;
      b0reg[e] = *(const unsigned int*)wp;        // row 2kk,  cols n,n+1
      b1reg[e] = *(const unsigned int*)(wp + N);  // row 2kk+1,cols n,n+1
    }
  };
  auto sttile = [&](int buf) {
    *(v4u*)&sA[buf][arow_ld][acol_ld] = areg;
#pragma unroll
    for (int e = 0; e < 4; e++) {
      const int u = tid + 256 * e;
      const int npair = u & 63;
      const int kk = u >> 6;
      // in-register 2x2 transpose of bf16 pairs
      const unsigned int d0 = (b0reg[e] & 0xffffu) | (b1reg[e] << 16);
      const unsigned int d1 = (b0reg[e] >> 16)     | (b1reg[e] & 0xffff0000u);
      *(unsigned int*)&sB[buf][2 * npair][2 * kk]     = d0;
      *(unsigned int*)&sB[buf][2 * npair + 1][2 * kk] = d1;
    }
  };

  ldtile(0);
  sttile(0);
  __syncthreads();

  int cur = 0;
  for (int k0 = 0; k0 < K; k0 += BKK) {
    const int kn = k0 + BKK;
    if (kn < K) ldtile(kn);  // next tile's global loads in flight under WMMA

    // Build fragments: lane half h holds k = 16h..16h+15 (same order for A and B).
    v16bf af[2], bfr[2];
#pragma unroll
    for (int ty = 0; ty < 2; ty++) {
      const int arow = wy * 32 + ty * 16 + l16;
#pragma unroll
      for (int j = 0; j < 8; j++) {
        const int kb = 16 * half + 2 * j;
        af[ty][2 * j]     = sA[cur][arow][kb];
        af[ty][2 * j + 1] = sA[cur][arow][kb + 1];
      }
    }
#pragma unroll
    for (int tx = 0; tx < 2; tx++) {
      const int bcol = wx * 32 + tx * 16 + l16;
#pragma unroll
      for (int j = 0; j < 8; j++) {   // contiguous 16 bf16 -> 2x ds_load_b128
        const int kb = 16 * half + 2 * j;
        bfr[tx][2 * j]     = sB[cur][bcol][kb];
        bfr[tx][2 * j + 1] = sB[cur][bcol][kb + 1];
      }
    }
#pragma unroll
    for (int ty = 0; ty < 2; ty++)
#pragma unroll
      for (int tx = 0; tx < 2; tx++)
        acc[ty][tx] = __builtin_amdgcn_wmma_f32_16x16x32_bf16(
            false, af[ty], false, bfr[tx], (short)0, acc[ty][tx], false, false);

    if (kn < K) sttile(cur ^ 1);
    __syncthreads();
    cur ^= 1;
  }

  // Epilogue. C layout: VGPR v holds row (v + 8*half) of a 16x16 tile, col = l16.
#pragma unroll
  for (int ty = 0; ty < 2; ty++) {
#pragma unroll
    for (int tx = 0; tx < 2; tx++) {
      const int col = bn + wx * 32 + tx * 16 + l16;
      const float bv = bias ? bias[col] : 0.0f;
#pragma unroll
      for (int v = 0; v < 8; v++) {
        const int m = bm + wy * 32 + ty * 16 + half * 8 + v;
        float val = acc[ty][tx][v] + bv;
        if (gelu) val = 0.5f * val * (1.0f + erff(val * 0.70710678118f));
        const long long off = (long long)m * N + col;
        if (resid) val += resid[off];
        if (Cf) Cf[off] = val;
        if (Cb) Cb[off] = (bf16_t)val;
      }
    }
  }
}

// ---------------------------------------------------------------- attention
// One 128-thread block per (query i, head h, batch b). qkv row = 2304 bf16:
// q [0,768) k [768,1536) v [1536,2304), head h at h*64 inside each third.
// bf16 pairs are loaded as dwords; bf16->f32 = <<16 bitcast.
__global__ void k_attn(const bf16_t* __restrict__ qkv, bf16_t* __restrict__ outb,
                       float* __restrict__ attn0, int n) {
  const int i = blockIdx.x, h = blockIdx.y, b = blockIdx.z;
  __shared__ float sq[DHEAD];
  __shared__ float sp[224];
  __shared__ float sred;
  const int t = threadIdx.x;  // 128
  const bf16_t* Q  = qkv + (long long)(b * n + i) * 2304 + h * DHEAD;
  const bf16_t* Kb = qkv + (long long)b * n * 2304 + DIMM + h * DHEAD;
  const bf16_t* Vb = qkv + (long long)b * n * 2304 + 2 * DIMM + h * DHEAD;
  if (t < DHEAD) sq[t] = (float)Q[t];
  __syncthreads();
  for (int j = t; j < n; j += 128) {
    const unsigned int* kr = (const unsigned int*)(Kb + (long long)j * 2304);
    float s = 0.f;
#pragma unroll
    for (int d2 = 0; d2 < 32; d2++) {
      const unsigned int u = kr[d2];
      s += sq[2 * d2]     * __uint_as_float(u << 16);
      s += sq[2 * d2 + 1] * __uint_as_float(u & 0xffff0000u);
    }
    sp[j] = s * 0.125f;  // DIM_HEAD^-0.5
  }
  __syncthreads();
  if (t == 0) { float m = -3.0e38f; for (int j = 0; j < n; j++) m = fmaxf(m, sp[j]); sred = m; }
  __syncthreads();
  const float mx = sred;
  for (int j = t; j < n; j += 128) sp[j] = __expf(sp[j] - mx);
  __syncthreads();
  if (t == 0) { float s = 0.f; for (int j = 0; j < n; j++) s += sp[j]; sred = 1.0f / s; }
  __syncthreads();
  const float inv = sred;
  for (int j = t; j < n; j += 128) sp[j] *= inv;
  __syncthreads();
  if (i == 0) {  // export CLS attention row for token pruning
    float* a0 = attn0 + (long long)(b * HEADS + h) * n;
    for (int j = t; j < n; j += 128) a0[j] = sp[j];
  }
  if (t < 32) {  // thread t owns dims (2t, 2t+1)
    const unsigned int* v32 = (const unsigned int*)Vb;
    float a0 = 0.f, a1 = 0.f;
    for (int j = 0; j < n; j++) {
      const unsigned int u = v32[(long long)j * 1152 + t];  // 2304 bf16 = 1152 dw
      const float pj = sp[j];
      a0 += pj * __uint_as_float(u << 16);
      a1 += pj * __uint_as_float(u & 0xffff0000u);
    }
    bf16_t* orow = outb + (long long)(b * n + i) * DIMM + h * DHEAD;
    orow[2 * t]     = (bf16_t)a0;
    orow[2 * t + 1] = (bf16_t)a1;
  }
}

// cls[b, j-1] = max over heads of attn0[b, h, j], j = 1..n-1
__global__ void k_clsmax(const float* __restrict__ attn0, float* __restrict__ cls, int n) {
  const int total = BATCH * (n - 1);
  const int gidx = blockIdx.x * blockDim.x + threadIdx.x;
  if (gidx >= total) return;
  const int b = gidx / (n - 1);
  const int j = gidx % (n - 1) + 1;
  float m = -3.0e38f;
  for (int h = 0; h < HEADS; h++)
    m = fmaxf(m, attn0[(long long)(b * HEADS + h) * n + j]);
  cls[gidx] = m;
}

// Per-batch serial top-k selection, then compact chosen indices ascending.
__global__ void k_topk(const float* __restrict__ cls, int* __restrict__ idx,
                       int nm1, int k) {
  const int b = threadIdx.x;
  if (b >= BATCH) return;
  const float* c = cls + (long long)b * nm1;
  bool chosen[256];
  for (int j = 0; j < nm1; j++) chosen[j] = false;
  for (int it = 0; it < k; it++) {
    float best = -3.0e38f; int bj = 0;
    for (int j = 0; j < nm1; j++) {
      const float v = c[j];
      if (!chosen[j] && v > best) { best = v; bj = j; }
    }
    chosen[bj] = true;
  }
  int* o = idx + (long long)b * (k + 1);
  o[0] = 0;
  int p = 1;
  for (int j = 0; j < nm1; j++) if (chosen[j]) o[p++] = j + 1;
}

// x_new[b,i,:] = emb[b,idx,:] + x_old[b,idx,:]
__global__ void k_gather_add(const float* __restrict__ emb, const float* __restrict__ xold,
                             const int* __restrict__ idx, float* __restrict__ xnew,
                             int nin, int nout) {
  const int row = blockIdx.x;  // BATCH * nout
  const int b = row / nout;
  const int i = row % nout;
  const int src = idx[(long long)b * nout + i];
  const float* e  = emb  + (long long)(b * nin + src) * DIMM;
  const float* xo = xold + (long long)(b * nin + src) * DIMM;
  float* xn = xnew + (long long)row * DIMM;
  const int t = threadIdx.x;  // 256
  xn[t]       = e[t]       + xo[t];
  xn[t + 256] = e[t + 256] + xo[t + 256];
  xn[t + 512] = e[t + 512] + xo[t + 512];
}

// ---------------------------------------------------------------- driver
extern "C" void kernel_launch(void* const* d_in, const int* in_sizes, int n_in,
                              void* d_out, int out_size, void* d_ws, size_t ws_size,
                              hipStream_t stream) {
  (void)in_sizes; (void)n_in; (void)out_size; (void)ws_size;
  const float* x_in   = (const float*)d_in[0];
  const float* ln1_g  = (const float*)d_in[1];
  const float* ln1_b  = (const float*)d_in[2];
  const float* w_qkv  = (const float*)d_in[3];
  const float* w_o    = (const float*)d_in[4];
  const float* b_o    = (const float*)d_in[5];
  const float* ln2_g  = (const float*)d_in[6];
  const float* ln2_b  = (const float*)d_in[7];
  const float* w1     = (const float*)d_in[8];
  const float* b1     = (const float*)d_in[9];
  const float* w2     = (const float*)d_in[10];
  const float* b2     = (const float*)d_in[11];
  const float* norm_g = (const float*)d_in[12];
  const float* norm_b = (const float*)d_in[13];
  static const int NP[DEPTHL] = {160, 128, 96, 64};  // static in reference

  // Workspace carve-out (256B aligned).
  char* p = (char*)d_ws;
  auto alloc = [&](size_t bytes) -> char* {
    char* r = p; p += (bytes + 255) & ~(size_t)255; return r;
  };
  bf16_t* wqkv_b = (bf16_t*)alloc((size_t)DEPTHL * DIMM * 3 * DIMM * 2);
  bf16_t* wo_b   = (bf16_t*)alloc((size_t)DEPTHL * DIMM * DIMM * 2);
  bf16_t* w1_b   = (bf16_t*)alloc((size_t)DEPTHL * DIMM * MLPD * 2);
  bf16_t* w2_b   = (bf16_t*)alloc((size_t)DEPTHL * MLPD * DIMM * 2);
  float*  x0     = (float*)alloc((size_t)BATCH * NTOK * DIMM * 4);
  float*  x1     = (float*)alloc((size_t)BATCH * NTOK * DIMM * 4);
  bf16_t* h_b    = (bf16_t*)alloc((size_t)BATCH * NTOK * DIMM * 2);
  bf16_t* qkv_b  = (bf16_t*)alloc((size_t)BATCH * NTOK * 3 * DIMM * 2);
  bf16_t* ao_b   = (bf16_t*)alloc((size_t)BATCH * NTOK * DIMM * 2);
  float*  emb_f  = (float*)alloc((size_t)BATCH * NTOK * DIMM * 4);
  bf16_t* f1_b   = (bf16_t*)alloc((size_t)BATCH * NTOK * MLPD * 2);
  float*  attn0  = (float*)alloc((size_t)BATCH * HEADS * NTOK * 4);
  float*  cls    = (float*)alloc((size_t)BATCH * (NTOK - 1) * 4);
  int*    idx    = (int*)alloc((size_t)BATCH * NTOK * 4);

  // Weights -> bf16 once per launch (vectorized, counts all divisible by 4).
  k_cvt_bf16<<<2048, 256, 0, stream>>>(w_qkv, wqkv_b, (long long)DEPTHL * DIMM * 3 * DIMM / 4);
  k_cvt_bf16<<<2048, 256, 0, stream>>>(w_o,   wo_b,   (long long)DEPTHL * DIMM * DIMM / 4);
  k_cvt_bf16<<<2048, 256, 0, stream>>>(w1,    w1_b,   (long long)DEPTHL * DIMM * MLPD / 4);
  k_cvt_bf16<<<2048, 256, 0, stream>>>(w2,    w2_b,   (long long)DEPTHL * MLPD * DIMM / 4);
  hipMemcpyAsync(x0, x_in, (size_t)BATCH * NTOK * DIMM * 4, hipMemcpyDeviceToDevice, stream);

  float* xc = x0;
  float* xn = x1;
  int nin = NTOK;
  for (int l = 0; l < DEPTHL; l++) {
    const int nout = NP[l] + 1;
    const int Min  = BATCH * nin;   // multiple of 64
    const int Mout = BATCH * nout;  // multiple of 64
    // LN1 -> bf16
    k_layernorm<<<Min, 256, 0, stream>>>(xc, ln1_g + l * DIMM, ln1_b + l * DIMM,
                                         h_b, (float*)nullptr);
    // QKV GEMM (bf16 out for attention)
    k_gemm_bf16<<<dim3((3 * DIMM) / BN, Min / BM), 256, 0, stream>>>(
        h_b, wqkv_b + (size_t)l * DIMM * 3 * DIMM, nullptr, nullptr,
        nullptr, qkv_b, Min, 3 * DIMM, DIMM, 0);
    // Attention + CLS row export
    k_attn<<<dim3(nin, HEADS, BATCH), 128, 0, stream>>>(qkv_b, ao_b, attn0, nin);
    // Pruning scores + top-k
    { const int tot = BATCH * (nin - 1);
      k_clsmax<<<(tot + 255) / 256, 256, 0, stream>>>(attn0, cls, nin); }
    k_topk<<<1, 64, 0, stream>>>(cls, idx, nin - 1, NP[l]);
    // Output projection (+bias) -> f32
    k_gemm_bf16<<<dim3(DIMM / BN, Min / BM), 256, 0, stream>>>(
        ao_b, wo_b + (size_t)l * DIMM * DIMM, b_o + l * DIMM, nullptr,
        emb_f, nullptr, Min, DIMM, DIMM, 0);
    // Gather kept tokens + residual
    k_gather_add<<<Mout, 256, 0, stream>>>(emb_f, xc, idx, xn, nin, nout);
    // LN2 -> bf16
    k_layernorm<<<Mout, 256, 0, stream>>>(xn, ln2_g + l * DIMM, ln2_b + l * DIMM,
                                          h_b, (float*)nullptr);
    // MLP up (+bias, exact GELU) -> bf16
    k_gemm_bf16<<<dim3(MLPD / BN, Mout / BM), 256, 0, stream>>>(
        h_b, w1_b + (size_t)l * DIMM * MLPD, b1 + l * MLPD, nullptr,
        nullptr, f1_b, Mout, MLPD, DIMM, 1);
    // MLP down (+bias, in-place residual) -> f32 x
    k_gemm_bf16<<<dim3(DIMM / BN, Mout / BM), 256, 0, stream>>>(
        f1_b, w2_b + (size_t)l * MLPD * DIMM, b2 + l * DIMM, xn,
        xn, nullptr, Mout, DIMM, MLPD, 0);
    float* tmp = xc; xc = xn; xn = tmp;
    nin = nout;
  }
  // Final LayerNorm -> fp32 output [B, 65, 768]
  k_layernorm<<<BATCH * nin, 256, 0, stream>>>(xc, norm_g, norm_b,
                                               (bf16_t*)nullptr, (float*)d_out);
}
